// DimeNet_88098369176151
// MI455X (gfx1250) — matrix operand order repaired and checked by reading, not compiled, not run on hardware
//
#include <hip/hip_runtime.h>

// ---------------- problem constants ----------------
#define HDIM    128
#define KBES    6
#define LMAX    7
#define LKDIM   42          // LMAX * KBES
#define NBLK    2
#define BILD    8
#define NTOUT   12
#define NEDGE   400000
#define NTRI    800000
#define NNODE   50000
#define NZB     13          // KBES + LMAX bessel zeros per order
#define MAT     (HDIM*HDIM) // 16384

#define FLAG_LOADC 1
#define FLAG_ACT   2

// ---------------- WMMA types (CDNA5 / gfx1250, wave32) ----------------
typedef __attribute__((ext_vector_type(16))) __bf16       v16bf;
typedef __attribute__((ext_vector_type(8)))  float        v8f;
typedef __attribute__((ext_vector_type(4)))  unsigned int u32x4;

union BFrag { v16bf v; u32x4 q[2]; };

// ---------------- small device helpers ----------------
__device__ __forceinline__ unsigned short f2bf(float f) {
  union { float f; unsigned u; } x; x.f = f;
  unsigned r = x.u + 0x7FFFu + ((x.u >> 16) & 1u);   // round-to-nearest-even
  return (unsigned short)(r >> 16);
}

__device__ __forceinline__ float silu_f(float x) {
  return x * (1.0f / (1.0f + __expf(-x)));
}

// envelope with P=6: 1/x + EA x^5 + EB x^6 + EC x^7, EA=-28 EB=48 EC=-21
__device__ __forceinline__ float envelope_f(float x) {
  float x2 = x * x, x4 = x2 * x2, x5 = x4 * x, x6 = x5 * x, x7 = x6 * x;
  return 1.0f / x - 28.0f * x5 + 48.0f * x6 - 21.0f * x7;
}

__device__ __forceinline__ float sph_jl_f(int l, float x) {
  float j0 = sinf(x) / x;
  if (l == 0) return j0;
  float j1 = sinf(x) / (x * x) - cosf(x) / x;
  float jm = j0, jc = j1;
  for (int m = 1; m < l; ++m) {
    float jn = (2.0f * m + 1.0f) / x * jc - jm;
    jm = jc; jc = jn;
  }
  return jc;
}

__device__ double jn_d(double x, int n) {
  double j0 = sin(x) / x;
  if (n == 0) return j0;
  double j1 = sin(x) / (x * x) - cos(x) / x;
  double jm = j0, jc = j1;
  for (int m = 1; m < n; ++m) {
    double t = (2.0 * m + 1.0) / x * jc - jm;
    jm = jc; jc = t;
  }
  return jc;
}

// ---------------- one-time tables: spherical Bessel zeros + normalizers ----
__global__ void bessel_tables_kernel(float* __restrict__ zln, float* __restrict__ nrm) {
  if (threadIdx.x != 0 || blockIdx.x != 0) return;
  const double PI_D = 3.14159265358979323846;
  double z[LMAX][NZB];
  for (int j = 0; j < NZB; ++j) z[0][j] = (j + 1) * PI_D;
  for (int l = 1; l < LMAX; ++l) {
    for (int j = 0; j < NZB - l; ++j) {
      double a = z[l - 1][j], b = z[l - 1][j + 1];
      for (int it = 0; it < 80; ++it) {
        double c = 0.5 * (a + b);
        if (jn_d(a, l) * jn_d(c, l) <= 0.0) b = c; else a = c;
      }
      z[l][j] = 0.5 * (a + b);
    }
  }
  for (int l = 0; l < LMAX; ++l)
    for (int k = 0; k < KBES; ++k) {
      zln[l * KBES + k] = (float)z[l][k];
      nrm[l * KBES + k] = (float)(sqrt(2.0) / fabs(jn_d(z[l][k], l + 1)));
    }
}

// ---------------- weight packing: f32 -> bf16 with arbitrary strides -------
// dst[n*Kd + k] = bf16(src[n*nStride + k*kStride])
__global__ void pack_bf16_kernel(unsigned short* __restrict__ dst,
                                 const float* __restrict__ src,
                                 int nRows, int Kd,
                                 long long nStride, long long kStride) {
  long long id = (long long)blockIdx.x * blockDim.x + threadIdx.x;
  if (id >= (long long)nRows * Kd) return;
  int n = (int)(id / Kd), k = (int)(id % Kd);
  dst[(long long)n * Kd + k] = f2bf(src[(long long)n * nStride + (long long)k * kStride]);
}

__global__ void cvt_bf16_kernel(unsigned short* __restrict__ dst,
                                const float* __restrict__ src, long long n) {
  long long i = (long long)blockIdx.x * blockDim.x + threadIdx.x;
  if (i < n) dst[i] = f2bf(src[i]);
}

__global__ void idx_kernel(const int* __restrict__ Z, const int* __restrict__ es,
                           const int* __restrict__ ed,
                           int* __restrict__ isrc, int* __restrict__ idst) {
  int e = blockIdx.x * blockDim.x + threadIdx.x;
  if (e >= NEDGE) return;
  isrc[e] = Z[es[e]];
  idst[e] = Z[ed[e]];
}

// ---------------- per-edge basis: rbf[6], rbf_env[42], rbf_p (bf16) -------
__global__ __launch_bounds__(128) void edge_init_kernel(
    const float* __restrict__ o, const float* __restrict__ zln,
    const float* __restrict__ nrm, const float* __restrict__ Wrbf,
    const float* __restrict__ brbf,
    float* __restrict__ rbf, float* __restrict__ envb,
    unsigned short* __restrict__ rbfp) {
  const int e = blockIdx.x;
  const int tid = threadIdx.x;
  const float PI_F = 3.14159265358979323846f;
  float ox = o[(long long)e * 3 + 0];
  float oy = o[(long long)e * 3 + 1];
  float oz = o[(long long)e * 3 + 2];
  float d = sqrtf(ox * ox + oy * oy + oz * oz);
  d = fmaxf(d, 1e-3f * 0.5f);
  float ds = d * 2.0f;                    // d / CUTOFF, CUTOFF = 0.5
  float envv = envelope_f(ds);
  __shared__ float srbf[KBES];
  if (tid < KBES) {
    float rv = envv * sinf(PI_F * (tid + 1) * ds);
    rbf[(long long)e * KBES + tid] = rv;
    srbf[tid] = rv;
  }
  if (tid < LKDIM) {
    int l = tid / KBES;
    float xx = ds * zln[tid];
    envb[(long long)e * LKDIM + tid] = envv * nrm[tid] * sph_jl_f(l, xx);
  }
  __syncthreads();
  float acc = brbf[tid];
  #pragma unroll
  for (int k = 0; k < KBES; ++k) acc += srbf[k] * Wrbf[k * HDIM + tid];
  rbfp[(long long)e * HDIM + tid] = f2bf(acc);
}

// ---------------- per-triplet: angles -> Legendre*coef, cbf[7] ------------
__global__ void cbf_kernel(const float* __restrict__ o, const int* __restrict__ lsrc,
                           const int* __restrict__ ldst, float* __restrict__ cbf) {
  int w = blockIdx.x * blockDim.x + threadIdx.x;
  if (w >= NTRI) return;
  int a = lsrc[w], b = ldst[w];
  float ax = o[(long long)a * 3], ay = o[(long long)a * 3 + 1], az = o[(long long)a * 3 + 2];
  float bx = o[(long long)b * 3], by = o[(long long)b * 3 + 1], bz = o[(long long)b * 3 + 2];
  float cx = ay * bz - az * by;
  float cy = az * bx - ax * bz;
  float cz = ax * by - ay * bx;
  float cn = sqrtf(cx * cx + cy * cy + cz * cz);
  float dt = ax * bx + ay * by + az * bz;
  float c  = cosf(atan2f(cn, dt));
  const float INV4PI = 0.07957747154594767f;
  float pm = 1.0f, pc = c;
  cbf[(long long)w * LMAX + 0] = pm * sqrtf(1.0f * INV4PI);
  cbf[(long long)w * LMAX + 1] = pc * sqrtf(3.0f * INV4PI);
  #pragma unroll
  for (int l = 1; l <= LMAX - 2; ++l) {
    float pn = ((2.0f * l + 1.0f) * c * pc - (float)l * pm) / (float)(l + 1);
    pm = pc; pc = pn;
    cbf[(long long)w * LMAX + (l + 1)] = pc * sqrtf((2.0f * (l + 1) + 1.0f) * INV4PI);
  }
}

// ---------------- per-triplet: sbf_p = (env[lg_src] * cbf) @ Wsbf[i] ------
__global__ void sbfp_kernel(const float* __restrict__ envb, const float* __restrict__ cbf,
                            const int* __restrict__ lsrc, const float* __restrict__ Wsbf,
                            float* __restrict__ sbfp) {
  int w = blockIdx.x * blockDim.x + threadIdx.x;
  if (w >= NTRI) return;
  const float* ep = envb + (long long)lsrc[w] * LKDIM;
  const float* cb = cbf + (long long)w * LMAX;
  float acc[BILD] = {0, 0, 0, 0, 0, 0, 0, 0};
  for (int l = 0; l < LMAX; ++l) {
    float cl = cb[l];
    #pragma unroll
    for (int k = 0; k < KBES; ++k) {
      float sv = ep[l * KBES + k] * cl;
      const float* wp = Wsbf + (l * KBES + k) * BILD;
      #pragma unroll
      for (int bq = 0; bq < BILD; ++bq) acc[bq] += sv * wp[bq];
    }
  }
  #pragma unroll
  for (int bq = 0; bq < BILD; ++bq) sbfp[(long long)w * BILD + bq] = acc[bq];
}

// ---------------- rbfw[e,h] = sum_k rbf[e,k] * W[k,h] ----------------------
__global__ __launch_bounds__(128) void rbf_proj_kernel(const float* __restrict__ rbf,
                                                       const float* __restrict__ W,
                                                       float* __restrict__ out) {
  int e = blockIdx.x, h = threadIdx.x;
  float acc = 0.f;
  #pragma unroll
  for (int k = 0; k < KBES; ++k) acc += rbf[(long long)e * KBES + k] * W[k * HDIM + h];
  out[(long long)e * HDIM + h] = acc;
}

// ---------------- out-block scatter: t[dst] += m * (rbf@Wrbf) -------------
__global__ __launch_bounds__(128) void out_scatter_kernel(const float* __restrict__ m,
                                                          const float* __restrict__ rbf,
                                                          const float* __restrict__ W,
                                                          const int* __restrict__ edst,
                                                          float* __restrict__ t) {
  int e = blockIdx.x, h = threadIdx.x;
  float acc = 0.f;
  #pragma unroll
  for (int k = 0; k < KBES; ++k) acc += rbf[(long long)e * KBES + k] * W[k * HDIM + h];
  float val = m[(long long)e * HDIM + h] * acc;
  atomicAdd(&t[(long long)edst[e] * HDIM + h], val);
}

// ---------------- generic WMMA GEMM, N = 128, K multiple of 32 ------------
// C[M,128] = act( gatherA[M,K] @ Bt[128,K]^T (+C) (+bias) ) (*emul) (+resid)
// one 16-row M strip per block; wave w owns cols [16w,16w+16)
__global__ __launch_bounds__(256) void gemm_wmma128(
    const unsigned short* __restrict__ A, const int* __restrict__ rowidx,
    const unsigned short* __restrict__ Bt, const float* __restrict__ bias,
    const float* __restrict__ emul, const float* __restrict__ resid,
    float* __restrict__ Cf, unsigned short* __restrict__ Cb,
    int Kdim, int flags) {
  const int lane  = threadIdx.x & 31;
  const int wave  = threadIdx.x >> 5;
  const int mBase = blockIdx.x << 4;
  const int nBase = wave << 4;
  const int kh    = lane >> 4;

  int aRow = mBase + (lane & 15);
  if (rowidx) aRow = rowidx[aRow];
  const unsigned short* Ap = A + (long long)aRow * Kdim + kh * 8;

  const int col = nBase + (lane & 15);
  const unsigned short* Bp = Bt + (long long)col * Kdim + kh * 8;

  v8f acc = {0.f, 0.f, 0.f, 0.f, 0.f, 0.f, 0.f, 0.f};
  for (int kb = 0; kb < Kdim; kb += 32) {
    BFrag a, b;
    a.q[0] = *(const u32x4*)(Ap + kb);
    a.q[1] = *(const u32x4*)(Ap + kb + 16);
    b.q[0] = *(const u32x4*)(Bp + kb);
    b.q[1] = *(const u32x4*)(Bp + kb + 16);
    acc = __builtin_amdgcn_wmma_f32_16x16x32_bf16(false, a.v, false, b.v,
                                                  (short)0, acc, false, false);
  }

  const int eRow0 = mBase + kh * 8;
  const float bcol = bias ? bias[col] : 0.f;
  #pragma unroll
  for (int v = 0; v < 8; ++v) {
    long long off = (long long)(eRow0 + v) * HDIM + col;
    float r = acc[v];
    if (flags & FLAG_LOADC) r += Cf[off];
    r += bcol;
    if (flags & FLAG_ACT) r = silu_f(r);
    if (emul)  r *= emul[off];
    if (resid) r += resid[off];
    Cf[off] = r;
    if (Cb) Cb[off] = f2bf(r);
  }
}

// ---------------- bilinear interaction (WMMA + gather A + scatter-add) ----
// msg[w,:] = sum_b sbfp[w,b] * (xkj[lg_src[w],:] @ BtBil[b]^T); x[lg_dst[w],:] += msg
__global__ __launch_bounds__(256) void bilinear_wmma(
    const unsigned short* __restrict__ xkj,
    const int* __restrict__ lsrc, const int* __restrict__ ldst,
    const unsigned short* __restrict__ BtBil,   // [8][128][128] bf16
    const float* __restrict__ sbfp,             // [T][8]
    float* __restrict__ xout) {
  const int lane  = threadIdx.x & 31;
  const int wave  = threadIdx.x >> 5;
  const int mBase = blockIdx.x << 4;
  const int nBase = wave << 4;
  const int kh    = lane >> 4;

  const int aTrip = mBase + (lane & 15);
  const int aRow  = lsrc[aTrip];
  const unsigned short* Ap = xkj + (long long)aRow * HDIM + kh * 8;
  const int col = nBase + (lane & 15);

  BFrag a[4];
  #pragma unroll
  for (int k4 = 0; k4 < 4; ++k4) {
    a[k4].q[0] = *(const u32x4*)(Ap + k4 * 32);
    a[k4].q[1] = *(const u32x4*)(Ap + k4 * 32 + 16);
  }

  const int eRow0 = mBase + kh * 8;
  int   dstRow[8];
  float sb[8][BILD];
  #pragma unroll
  for (int v = 0; v < 8; ++v) {
    dstRow[v] = ldst[eRow0 + v];
    const float* sp = sbfp + (long long)(eRow0 + v) * BILD;
    #pragma unroll
    for (int bq = 0; bq < BILD; ++bq) sb[v][bq] = sp[bq];
  }

  float msg[8] = {0, 0, 0, 0, 0, 0, 0, 0};
  for (int bq = 0; bq < BILD; ++bq) {
    const unsigned short* Bp = BtBil + ((long long)bq * HDIM + col) * HDIM + kh * 8;
    v8f t = {0.f, 0.f, 0.f, 0.f, 0.f, 0.f, 0.f, 0.f};
    #pragma unroll
    for (int k4 = 0; k4 < 4; ++k4) {
      BFrag b;
      b.q[0] = *(const u32x4*)(Bp + k4 * 32);
      b.q[1] = *(const u32x4*)(Bp + k4 * 32 + 16);
      t = __builtin_amdgcn_wmma_f32_16x16x32_bf16(false, a[k4].v, false, b.v,
                                                  (short)0, t, false, false);
    }
    #pragma unroll
    for (int v = 0; v < 8; ++v) msg[v] += sb[v][bq] * t[v];
  }
  #pragma unroll
  for (int v = 0; v < 8; ++v)
    atomicAdd(&xout[(long long)dstRow[v] * HDIM + col], msg[v]);
}

// ---------------- graph readout: P += sum_nodes t @ Wout ------------------
__global__ void readout_kernel(const float* __restrict__ t, const float* __restrict__ Wout,
                               float* __restrict__ Pacc, int nNodes) {
  int node = blockIdx.x * blockDim.x + threadIdx.x;
  if (node >= nNodes) return;
  const float* tr = t + (long long)node * HDIM;
  float acc[NTOUT];
  #pragma unroll
  for (int j = 0; j < NTOUT; ++j) acc[j] = 0.f;
  for (int k = 0; k < HDIM; ++k) {
    float tv = tr[k];
    const float* w = Wout + k * NTOUT;
    #pragma unroll
    for (int j = 0; j < NTOUT; ++j) acc[j] += tv * w[j];
  }
  #pragma unroll
  for (int j = 0; j < NTOUT; ++j) atomicAdd(&Pacc[j], acc[j]);
}

__global__ void finalize_kernel(const float* __restrict__ Pacc, float* __restrict__ out) {
  if (threadIdx.x < NTOUT) out[threadIdx.x] = Pacc[threadIdx.x];
}

// =========================================================================
extern "C" void kernel_launch(void* const* d_in, const int* in_sizes, int n_in,
                              void* d_out, int out_size, void* d_ws, size_t ws_size,
                              hipStream_t stream) {
  const int*   Z         = (const int*)d_in[0];
  const int*   edge_src  = (const int*)d_in[1];
  const int*   edge_dst  = (const int*)d_in[2];
  const int*   lg_src    = (const int*)d_in[3];
  const int*   lg_dst    = (const int*)d_in[4];
  const float* o         = (const float*)d_in[5];
  /* d_in[6] = freq, recomputed inline */
  const float* emb_table = (const float*)d_in[7];
  const float* emb_Wrbf  = (const float*)d_in[8];
  const float* emb_brbf  = (const float*)d_in[9];
  const float* emb_Wd    = (const float*)d_in[10];
  const float* emb_bd    = (const float*)d_in[11];
  const float* ib_Wrbf   = (const float*)d_in[12];
  const float* ib_Wsbf   = (const float*)d_in[13];
  const float* ib_Wji    = (const float*)d_in[14];
  const float* ib_bji    = (const float*)d_in[15];
  const float* ib_Wkj    = (const float*)d_in[16];
  const float* ib_bkj    = (const float*)d_in[17];
  const float* ib_Wbilin = (const float*)d_in[18];
  const float* ib_bsW1   = (const float*)d_in[19];
  const float* ib_bsb1   = (const float*)d_in[20];
  const float* ib_bsW2   = (const float*)d_in[21];
  const float* ib_bsb2   = (const float*)d_in[22];
  const float* ib_fbsW   = (const float*)d_in[23];
  const float* ib_fbsb   = (const float*)d_in[24];
  const float* ib_asW1   = (const float*)d_in[25];
  const float* ib_asb1   = (const float*)d_in[26];
  const float* ib_asW2   = (const float*)d_in[27];
  const float* ib_asb2   = (const float*)d_in[28];
  const float* ob_Wrbf   = (const float*)d_in[29];
  const float* ob_Wd     = (const float*)d_in[30];
  const float* ob_bd     = (const float*)d_in[31];
  const float* ob_Wout   = (const float*)d_in[32];
  (void)in_sizes; (void)n_in; (void)out_size; (void)ws_size;

  // ---- workspace carve-up ----
  char* ws = (char*)d_ws;
  size_t cur = 0;
  auto alloc = [&](size_t bytes) -> char* {
    cur = (cur + 255) & ~(size_t)255;
    char* p = ws + cur; cur += bytes; return p;
  };
  float* zln  = (float*)alloc(LKDIM * 4);
  float* nrm  = (float*)alloc(LKDIM * 4);
  unsigned short* embtab_bf = (unsigned short*)alloc((size_t)95 * HDIM * 2);
  unsigned short* bt_emb = (unsigned short*)alloc((size_t)3 * MAT * 2);
  unsigned short* bt_ji  = (unsigned short*)alloc((size_t)NBLK * MAT * 2);
  unsigned short* bt_kj  = (unsigned short*)alloc((size_t)NBLK * MAT * 2);
  unsigned short* bt_bs1 = (unsigned short*)alloc((size_t)NBLK * MAT * 2);
  unsigned short* bt_bs2 = (unsigned short*)alloc((size_t)NBLK * MAT * 2);
  unsigned short* bt_fbs = (unsigned short*)alloc((size_t)NBLK * MAT * 2);
  unsigned short* bt_as1 = (unsigned short*)alloc((size_t)NBLK * 2 * MAT * 2);
  unsigned short* bt_as2 = (unsigned short*)alloc((size_t)NBLK * 2 * MAT * 2);
  unsigned short* bt_bil = (unsigned short*)alloc((size_t)NBLK * BILD * MAT * 2);
  unsigned short* bt_obd = (unsigned short*)alloc((size_t)(NBLK + 1) * 3 * MAT * 2);
  int*   idx_src = (int*)alloc((size_t)NEDGE * 4);
  int*   idx_dst = (int*)alloc((size_t)NEDGE * 4);
  float* rbf     = (float*)alloc((size_t)NEDGE * KBES * 4);
  float* envb    = (float*)alloc((size_t)NEDGE * LKDIM * 4);
  unsigned short* rbfp_bf = (unsigned short*)alloc((size_t)NEDGE * HDIM * 2);
  float* m_f   = (float*)alloc((size_t)NEDGE * HDIM * 4);
  unsigned short* m_bf = (unsigned short*)alloc((size_t)NEDGE * HDIM * 2);
  float* x_f   = (float*)alloc((size_t)NEDGE * HDIM * 4);
  unsigned short* x_bf = (unsigned short*)alloc((size_t)NEDGE * HDIM * 2);
  float* u_f   = (float*)alloc((size_t)NEDGE * HDIM * 4);
  unsigned short* u_bf = (unsigned short*)alloc((size_t)NEDGE * HDIM * 2);
  unsigned short* xkj_bf = (unsigned short*)alloc((size_t)NEDGE * HDIM * 2);
  float* rbfw  = (float*)alloc((size_t)NEDGE * HDIM * 4);
  float* cbf   = (float*)alloc((size_t)NTRI * LMAX * 4);
  float* sbfp  = (float*)alloc((size_t)NTRI * BILD * 4);
  float* tA    = (float*)alloc((size_t)NNODE * HDIM * 4);
  unsigned short* tA_bf = (unsigned short*)alloc((size_t)NNODE * HDIM * 2);
  float* tB    = (float*)alloc((size_t)NNODE * HDIM * 4);
  unsigned short* tB_bf = (unsigned short*)alloc((size_t)NNODE * HDIM * 2);
  float* Pacc  = (float*)alloc(64);

  auto pack = [&](unsigned short* dst, const float* src, int nRows, int Kd,
                  long long nStride, long long kStride) {
    int total = nRows * Kd;
    pack_bf16_kernel<<<(total + 255) / 256, 256, 0, stream>>>(dst, src, nRows, Kd,
                                                              nStride, kStride);
  };

  // ---- tables + weight packing (bf16, transposed for WMMA-B) ----
  bessel_tables_kernel<<<1, 1, 0, stream>>>(zln, nrm);
  pack(embtab_bf, emb_table, 95, HDIM, HDIM, 1);                // plain convert
  for (int p = 0; p < 3; ++p)                                   // concat-GEMM split
    pack(bt_emb + (size_t)p * MAT, emb_Wd + (size_t)p * MAT, HDIM, HDIM, 1, HDIM);
  for (int i = 0; i < NBLK; ++i) {
    pack(bt_ji  + (size_t)i * MAT, ib_Wji  + (size_t)i * MAT, HDIM, HDIM, 1, HDIM);
    pack(bt_kj  + (size_t)i * MAT, ib_Wkj  + (size_t)i * MAT, HDIM, HDIM, 1, HDIM);
    pack(bt_bs1 + (size_t)i * MAT, ib_bsW1 + (size_t)i * MAT, HDIM, HDIM, 1, HDIM);
    pack(bt_bs2 + (size_t)i * MAT, ib_bsW2 + (size_t)i * MAT, HDIM, HDIM, 1, HDIM);
    pack(bt_fbs + (size_t)i * MAT, ib_fbsW + (size_t)i * MAT, HDIM, HDIM, 1, HDIM);
    for (int j = 0; j < 2; ++j) {
      pack(bt_as1 + (size_t)(i * 2 + j) * MAT, ib_asW1 + (size_t)(i * 2 + j) * MAT,
           HDIM, HDIM, 1, HDIM);
      pack(bt_as2 + (size_t)(i * 2 + j) * MAT, ib_asW2 + (size_t)(i * 2 + j) * MAT,
           HDIM, HDIM, 1, HDIM);
    }
    // Wbilin[i, n, b, k] -> Bt[b][n][k]: already n-major with stride BILD*HDIM
    for (int b = 0; b < BILD; ++b)
      pack(bt_bil + (size_t)(i * BILD + b) * MAT,
           ib_Wbilin + (size_t)i * HDIM * BILD * HDIM + (size_t)b * HDIM,
           HDIM, HDIM, (long long)BILD * HDIM, 1);
  }
  for (int i = 0; i < NBLK + 1; ++i)
    for (int j = 0; j < 3; ++j)
      pack(bt_obd + (size_t)(i * 3 + j) * MAT, ob_Wd + (size_t)(i * 3 + j) * MAT,
           HDIM, HDIM, 1, HDIM);

  // ---- per-edge / per-triplet basis ----
  idx_kernel<<<(NEDGE + 255) / 256, 256, 0, stream>>>(Z, edge_src, edge_dst,
                                                      idx_src, idx_dst);
  edge_init_kernel<<<NEDGE, 128, 0, stream>>>(o, zln, nrm, emb_Wrbf, emb_brbf,
                                              rbf, envb, rbfp_bf);
  cbf_kernel<<<(NTRI + 255) / 256, 256, 0, stream>>>(o, lg_src, lg_dst, cbf);

  const int GE = NEDGE / 16, GN = NNODE / 16, GT = NTRI / 16;

  // ---- embedding block: m = silu([h_src, h_dst, rbf_p] @ Wd + bd) ----
  gemm_wmma128<<<GE, 256, 0, stream>>>(embtab_bf, idx_src, bt_emb, nullptr,
                                       nullptr, nullptr, m_f, nullptr, HDIM, 0);
  gemm_wmma128<<<GE, 256, 0, stream>>>(embtab_bf, idx_dst, bt_emb + MAT, nullptr,
                                       nullptr, nullptr, m_f, nullptr, HDIM, FLAG_LOADC);
  gemm_wmma128<<<GE, 256, 0, stream>>>(rbfp_bf, nullptr, bt_emb + 2 * MAT, emb_bd,
                                       nullptr, nullptr, m_f, m_bf, HDIM,
                                       FLAG_LOADC | FLAG_ACT);

  hipMemsetAsync(Pacc, 0, 64, stream);

  auto out_block = [&](int i) {
    hipMemsetAsync(tA, 0, (size_t)NNODE * HDIM * 4, stream);
    out_scatter_kernel<<<NEDGE, 128, 0, stream>>>(m_f, rbf,
                                                  ob_Wrbf + (size_t)i * KBES * HDIM,
                                                  edge_dst, tA);
    cvt_bf16_kernel<<<((size_t)NNODE * HDIM + 255) / 256, 256, 0, stream>>>(
        tA_bf, tA, (long long)NNODE * HDIM);
    gemm_wmma128<<<GN, 256, 0, stream>>>(tA_bf, nullptr, bt_obd + (size_t)(i * 3 + 0) * MAT,
                                         ob_bd + (size_t)(i * 3 + 0) * HDIM,
                                         nullptr, nullptr, tB, tB_bf, HDIM, FLAG_ACT);
    gemm_wmma128<<<GN, 256, 0, stream>>>(tB_bf, nullptr, bt_obd + (size_t)(i * 3 + 1) * MAT,
                                         ob_bd + (size_t)(i * 3 + 1) * HDIM,
                                         nullptr, nullptr, tA, tA_bf, HDIM, FLAG_ACT);
    gemm_wmma128<<<GN, 256, 0, stream>>>(tA_bf, nullptr, bt_obd + (size_t)(i * 3 + 2) * MAT,
                                         ob_bd + (size_t)(i * 3 + 2) * HDIM,
                                         nullptr, nullptr, tB, tB_bf, HDIM, FLAG_ACT);
    readout_kernel<<<(NNODE + 255) / 256, 256, 0, stream>>>(
        tB, ob_Wout + (size_t)i * HDIM * NTOUT, Pacc, NNODE);
  };

  out_block(0);

  // ---- interaction blocks ----
  for (int i = 0; i < NBLK; ++i) {
    // x_ji
    gemm_wmma128<<<GE, 256, 0, stream>>>(m_bf, nullptr, bt_ji + (size_t)i * MAT,
                                         ib_bji + (size_t)i * HDIM,
                                         nullptr, nullptr, x_f, nullptr, HDIM, FLAG_ACT);
    // x_kj = silu(m@Wkj+b) * (rbf@Wrbf)
    rbf_proj_kernel<<<NEDGE, 128, 0, stream>>>(rbf, ib_Wrbf + (size_t)i * KBES * HDIM, rbfw);
    gemm_wmma128<<<GE, 256, 0, stream>>>(m_bf, nullptr, bt_kj + (size_t)i * MAT,
                                         ib_bkj + (size_t)i * HDIM,
                                         rbfw, nullptr, u_f, xkj_bf, HDIM, FLAG_ACT);
    // sbf projection + bilinear message + segment_sum into x
    sbfp_kernel<<<(NTRI + 255) / 256, 256, 0, stream>>>(envb, cbf, lg_src,
                                                        ib_Wsbf + (size_t)i * LKDIM * BILD,
                                                        sbfp);
    bilinear_wmma<<<GT, 256, 0, stream>>>(xkj_bf, lg_src, lg_dst,
                                          bt_bil + (size_t)i * BILD * MAT, sbfp, x_f);
    cvt_bf16_kernel<<<((size_t)NEDGE * HDIM + 255) / 256, 256, 0, stream>>>(
        x_bf, x_f, (long long)NEDGE * HDIM);
    // x = x + silu(silu(x@bsW1+b)@bsW2+b)
    gemm_wmma128<<<GE, 256, 0, stream>>>(x_bf, nullptr, bt_bs1 + (size_t)i * MAT,
                                         ib_bsb1 + (size_t)i * HDIM,
                                         nullptr, nullptr, u_f, u_bf, HDIM, FLAG_ACT);
    gemm_wmma128<<<GE, 256, 0, stream>>>(u_bf, nullptr, bt_bs2 + (size_t)i * MAT,
                                         ib_bsb2 + (size_t)i * HDIM,
                                         nullptr, x_f, x_f, x_bf, HDIM, FLAG_ACT);
    // m = m + silu(x@fbsW+b)
    gemm_wmma128<<<GE, 256, 0, stream>>>(x_bf, nullptr, bt_fbs + (size_t)i * MAT,
                                         ib_fbsb + (size_t)i * HDIM,
                                         nullptr, m_f, m_f, m_bf, HDIM, FLAG_ACT);
    // 2x residual MLPs on m
    for (int j = 0; j < 2; ++j) {
      gemm_wmma128<<<GE, 256, 0, stream>>>(m_bf, nullptr,
                                           bt_as1 + (size_t)(i * 2 + j) * MAT,
                                           ib_asb1 + (size_t)(i * 2 + j) * HDIM,
                                           nullptr, nullptr, u_f, u_bf, HDIM, FLAG_ACT);
      gemm_wmma128<<<GE, 256, 0, stream>>>(u_bf, nullptr,
                                           bt_as2 + (size_t)(i * 2 + j) * MAT,
                                           ib_asb2 + (size_t)(i * 2 + j) * HDIM,
                                           nullptr, m_f, m_f, m_bf, HDIM, FLAG_ACT);
    }
    out_block(i + 1);
  }

  finalize_kernel<<<1, 32, 0, stream>>>(Pacc, (float*)d_out);
}
